// Bottleneck_61022895341826
// MI455X (gfx1250) — 8-bit matrix operand order repaired and checked by reading, compiled, not run on hardware
//
#include <hip/hip_runtime.h>
#include <hip/hip_bf16.h>
#include <stdint.h>

typedef __attribute__((ext_vector_type(8))) int v8i;

#define QF    127.0f
#define INVQ  (1.0f/127.0f)
#define INVQ2 (1.0f/16129.0f)

#define LDA 80
#define LDB 80

__device__ __forceinline__ void async_copy_b128(void* lds_ptr, const void* gptr) {
    unsigned lds_off = (unsigned)(uintptr_t)lds_ptr;
    unsigned long long ga = (unsigned long long)(uintptr_t)gptr;
    asm volatile("global_load_async_to_lds_b128 %0, %1, off"
                 :: "v"(lds_off), "v"(ga) : "memory");
}
__device__ __forceinline__ void wait_async0() {
    asm volatile("s_wait_asynccnt 0x0" ::: "memory");
}

__global__ void init_scales_kernel(float* s) {
    if (threadIdx.x < 4) s[threadIdx.x] = 0.0f;
}

__global__ void absmax_kernel(const float* __restrict__ w, int n,
                              unsigned int* __restrict__ slot) {
    __shared__ float red[256];
    int tid = threadIdx.x;
    float m = 0.0f;
    for (int i = blockIdx.x * blockDim.x + tid; i < n; i += gridDim.x * blockDim.x)
        m = fmaxf(m, fabsf(w[i]));
    red[tid] = m;
    __syncthreads();
    for (int s = 128; s > 0; s >>= 1) {
        if (tid < s) red[tid] = fmaxf(red[tid], red[tid + s]);
        __syncthreads();
    }
    if (tid == 0) atomicMax(slot, __float_as_uint(red[0]));
}

__device__ inline unsigned pack4q(float q0, float q1, float q2, float q3) {
    unsigned b0 = (unsigned)(unsigned char)(signed char)(int)q0;
    unsigned b1 = (unsigned)(unsigned char)(signed char)(int)q1;
    unsigned b2 = (unsigned)(unsigned char)(signed char)(int)q2;
    unsigned b3 = (unsigned)(unsigned char)(signed char)(int)q3;
    return b0 | (b1 << 8) | (b2 << 16) | (b3 << 24);
}

__device__ inline float fq127(float v) {
    return rintf(fminf(fmaxf(v, -1.0f), 1.0f) * QF);
}

__global__ void quantw_kernel(const float* __restrict__ w,
                              const float* __restrict__ s,
                              unsigned* __restrict__ wq, int n4) {
    int i = blockIdx.x * blockDim.x + threadIdx.x;
    if (i >= n4) return;
    float inv = 1.0f / (*s);
    float4 f = ((const float4*)w)[i];
    wq[i] = pack4q(fq127(f.x * inv), fq127(f.y * inv), fq127(f.z * inv), fq127(f.w * inv));
}

__global__ void quantx_kernel(const float* __restrict__ x,
                              unsigned* __restrict__ xq, int n4) {
    int i = blockIdx.x * blockDim.x + threadIdx.x;
    if (i >= n4) return;
    float4 f = ((const float4*)x)[i];
    xq[i] = pack4q(fq127(f.x * 2.0f), fq127(f.y * 2.0f), fq127(f.z * 2.0f), fq127(f.w * 2.0f));
}

__device__ inline v8i load_a_frag(const signed char (*lA)[LDA], int mloc, int lane) {
    int kh = (lane >> 4) << 3;
    const signed char* r = lA[mloc];
    v8i a;
#pragma unroll
    for (int g = 0; g < 4; ++g) {
        uint2 d = *(const uint2*)(r + g * 16 + kh);
        a[2 * g]     = (int)d.x;
        a[2 * g + 1] = (int)d.y;
    }
    return a;
}

__device__ inline v8i load_b_frag(const signed char (*lB)[LDB], int nw0, int lane) {
    const int n = nw0 + (lane & 15);
    const int kh = (lane >> 4) << 3;
    v8i b;
#pragma unroll
    for (int i = 0; i < 8; ++i) {
        unsigned w = 0;
#pragma unroll
        for (int j = 0; j < 4; ++j) {
            int s = 4 * i + j;
            int k = 16 * (s >> 3) + kh + (s & 7);
            w |= (unsigned)(unsigned char)lB[k][n] << (8 * j);
        }
        b[i] = (int)w;
    }
    return b;
}

#define WMMA_IU8(A, B, C) __builtin_amdgcn_wmma_i32_16x16x64_iu8(true, A, true, B, C, false, false)

__global__ __launch_bounds__(256) void conv1_kernel(
    const signed char* __restrict__ xq, const signed char* __restrict__ w1q,
    const float* __restrict__ b1, const float* __restrict__ scales,
    signed char* __restrict__ a2q) {
    __shared__ signed char lA[64][LDA];
    __shared__ signed char lB[64][LDB];
    const int tid = threadIdx.x, lane = tid & 31, wv = tid >> 5;
    const int wm = wv & 3, wn = wv >> 2;
    const int n0 = blockIdx.x * 64, m0 = blockIdx.y * 64, bb = blockIdx.z;
    const int K = 512, NPB = 784;
    const signed char* xb = xq + (size_t)bb * K * NPB;
    const int row = tid >> 2, seg = (tid & 3) * 16;
    v8i acc0 = {0,0,0,0,0,0,0,0};
    v8i acc1 = {0,0,0,0,0,0,0,0};

    for (int k0 = 0; k0 < K; k0 += 64) {
        async_copy_b128(lA[row] + seg, w1q + (size_t)(m0 + row) * K + k0 + seg);
        async_copy_b128(lB[row] + seg, xb + (size_t)(k0 + row) * NPB + n0 + seg);
        if (k0 + 64 < K)
            __builtin_prefetch(w1q + (size_t)(m0 + row) * K + k0 + 64 + seg, 0, 3);
        wait_async0();
        __syncthreads();
        v8i a  = load_a_frag(lA, (wm << 4) + (lane & 15), lane);
        v8i bf0 = load_b_frag(lB, (wn << 5), lane);
        v8i bf1 = load_b_frag(lB, (wn << 5) + 16, lane);
        acc0 = WMMA_IU8(a, bf0, acc0);
        acc1 = WMMA_IU8(a, bf1, acc1);
        __syncthreads();
    }

    const float c1s = scales[0];
    const int mb = m0 + (wm << 4) + ((lane >> 4) << 3);
    signed char* dstb = a2q + (size_t)bb * 256 * NPB;
#pragma unroll
    for (int s = 0; s < 2; ++s) {
        const v8i& acc = s ? acc1 : acc0;
        int n = n0 + (wn << 5) + (s << 4) + (lane & 15);
        if (n < NPB) {
#pragma unroll
            for (int v = 0; v < 8; ++v) {
                int m = mb + v;
                float bias = rintf(b1[m] * QF) * INVQ / (c1s * 0.5f);
                float t = ((float)acc[v] * INVQ2 + bias) * c1s;
                t = fminf(fmaxf(t * 2.0f, 0.0f), 6.0f) * 0.5f;
                float q = rintf(fminf(fmaxf(t * 2.0f, -1.0f), 1.0f) * QF);
                dstb[(size_t)m * NPB + n] = (signed char)(int)q;
            }
        }
    }
}

__global__ __launch_bounds__(256) void conv2_kernel(
    const signed char* __restrict__ a2q, const signed char* __restrict__ w2q,
    const float* __restrict__ b2, const float* __restrict__ scales,
    signed char* __restrict__ a3q) {
    __shared__ signed char lA[64][LDA];
    __shared__ signed char lB[64][LDB];
    const int tid = threadIdx.x, lane = tid & 31, wv = tid >> 5;
    const int wm = wv & 3, wn = wv >> 2;
    const int n0 = blockIdx.x * 64, m0 = blockIdx.y * 64, bb = blockIdx.z;
    const int K = 2304, NPB = 196;
    const int row = tid >> 2, seg = (tid & 3) * 16;
    v8i acc0 = {0,0,0,0,0,0,0,0};
    v8i acc1 = {0,0,0,0,0,0,0,0};

    for (int k0 = 0; k0 < K; k0 += 64) {
        async_copy_b128(lA[row] + seg, w2q + (size_t)(m0 + row) * K + k0 + seg);
        if (k0 + 64 < K)
            __builtin_prefetch(w2q + (size_t)(m0 + row) * K + k0 + 64 + seg, 0, 3);
        {
            union { uint4 v; signed char c[16]; } t;
            int k = k0 + row;
            int c = k / 9, r = k - 9 * c;
            int dy = r / 3, dx = r - 3 * dy;
            const signed char* src = a2q + ((size_t)bb * 256 + c) * 784;
            int nn = n0 + seg;
            int yy = nn / 14, xx = nn - 14 * yy;
#pragma unroll
            for (int j = 0; j < 16; ++j) {
                int iy = 2 * yy + dy - 1, ix = 2 * xx + dx - 1;
                bool ok = (nn < NPB) & ((unsigned)iy < 28u) & ((unsigned)ix < 28u);
                signed char val = src[ok ? iy * 28 + ix : 0];
                t.c[j] = ok ? val : (signed char)0;
                ++nn; ++xx;
                if (xx == 14) { xx = 0; ++yy; }
            }
            *(uint4*)(lB[row] + seg) = t.v;
        }
        wait_async0();
        __syncthreads();
        v8i a  = load_a_frag(lA, (wm << 4) + (lane & 15), lane);
        v8i bf0 = load_b_frag(lB, (wn << 5), lane);
        v8i bf1 = load_b_frag(lB, (wn << 5) + 16, lane);
        acc0 = WMMA_IU8(a, bf0, acc0);
        acc1 = WMMA_IU8(a, bf1, acc1);
        __syncthreads();
    }

    const float c2s = scales[1];
    const int mb = m0 + (wm << 4) + ((lane >> 4) << 3);
    signed char* dstb = a3q + (size_t)bb * 256 * NPB;
#pragma unroll
    for (int s = 0; s < 2; ++s) {
        const v8i& acc = s ? acc1 : acc0;
        int n = n0 + (wn << 5) + (s << 4) + (lane & 15);
        if (n < NPB) {
#pragma unroll
            for (int v = 0; v < 8; ++v) {
                int m = mb + v;
                float bias = rintf(b2[m] * QF) * INVQ / (c2s * 0.5f);
                float t = ((float)acc[v] * INVQ2 + bias) * c2s;
                t = fminf(fmaxf(t * 2.0f, 0.0f), 6.0f) * 0.5f;
                float q = rintf(fminf(fmaxf(t * 2.0f, -1.0f), 1.0f) * QF);
                dstb[(size_t)m * NPB + n] = (signed char)(int)q;
            }
        }
    }
}

__global__ __launch_bounds__(256) void conv3s_kernel(
    const signed char* __restrict__ a3q, const signed char* __restrict__ xq,
    const signed char* __restrict__ w3q, const signed char* __restrict__ wsq,
    const float* __restrict__ b3, const float* __restrict__ bs,
    const float* __restrict__ scales, float* __restrict__ out) {
    __shared__ signed char lA[64][LDA];
    __shared__ signed char lB[64][LDB];
    const int tid = threadIdx.x, lane = tid & 31, wv = tid >> 5;
    const int wm = wv & 3, wn = wv >> 2;
    const int n0 = blockIdx.x * 64, m0 = blockIdx.y * 64, bb = blockIdx.z;
    const int NPB = 196;
    const int row = tid >> 2, seg = (tid & 3) * 16;
    v8i acc30 = {0,0,0,0,0,0,0,0};
    v8i acc31 = {0,0,0,0,0,0,0,0};
    v8i accS0 = {0,0,0,0,0,0,0,0};
    v8i accS1 = {0,0,0,0,0,0,0,0};

    for (int k0 = 0; k0 < 256; k0 += 64) {
        async_copy_b128(lA[row] + seg, w3q + (size_t)(m0 + row) * 256 + k0 + seg);
        {
            const signed char* src = a3q + ((size_t)bb * 256 + k0 + row) * NPB + n0 + seg;
            uint4 t;
            t.x = *(const unsigned*)(src + 0);
            t.y = *(const unsigned*)(src + 4);
            t.z = *(const unsigned*)(src + 8);
            t.w = *(const unsigned*)(src + 12);
            *(uint4*)(lB[row] + seg) = t;
        }
        wait_async0();
        __syncthreads();
        v8i a  = load_a_frag(lA, (wm << 4) + (lane & 15), lane);
        v8i bf0 = load_b_frag(lB, (wn << 5), lane);
        v8i bf1 = load_b_frag(lB, (wn << 5) + 16, lane);
        acc30 = WMMA_IU8(a, bf0, acc30);
        acc31 = WMMA_IU8(a, bf1, acc31);
        __syncthreads();
    }
    for (int k0 = 0; k0 < 512; k0 += 64) {
        async_copy_b128(lA[row] + seg, wsq + (size_t)(m0 + row) * 512 + k0 + seg);
        if (k0 + 64 < 512)
            __builtin_prefetch(wsq + (size_t)(m0 + row) * 512 + k0 + 64 + seg, 0, 3);
        {
            union { uint4 v; signed char c[16]; } t;
            const signed char* src = xq + ((size_t)bb * 512 + k0 + row) * 784;
            int nn = n0 + seg;
            int yy = nn / 14, xx = nn - 14 * yy;
#pragma unroll
            for (int j = 0; j < 16; ++j) {
                bool ok = nn < NPB;
                signed char val = src[ok ? yy * 56 + 2 * xx : 0];
                t.c[j] = ok ? val : (signed char)0;
                ++nn; ++xx;
                if (xx == 14) { xx = 0; ++yy; }
            }
            *(uint4*)(lB[row] + seg) = t.v;
        }
        wait_async0();
        __syncthreads();
        v8i a  = load_a_frag(lA, (wm << 4) + (lane & 15), lane);
        v8i bf0 = load_b_frag(lB, (wn << 5), lane);
        v8i bf1 = load_b_frag(lB, (wn << 5) + 16, lane);
        accS0 = WMMA_IU8(a, bf0, accS0);
        accS1 = WMMA_IU8(a, bf1, accS1);
        __syncthreads();
    }

    const float c2s = scales[1], c3s = scales[2], css = scales[3];
    const int mb = m0 + (wm << 4) + ((lane >> 4) << 3);
    float* dstb = out + (size_t)bb * 1024 * NPB;
#pragma unroll
    for (int s = 0; s < 2; ++s) {
        const v8i& a3 = s ? acc31 : acc30;
        const v8i& aS = s ? accS1 : accS0;
        int n = n0 + (wn << 5) + (s << 4) + (lane & 15);
        if (n < NPB) {
#pragma unroll
            for (int v = 0; v < 8; ++v) {
                int m = mb + v;
                float b3q = rintf(b3[m] * QF) * INVQ / (c2s * 0.5f);
                float t3 = ((float)a3[v] * INVQ2 + b3q) * c3s * 0.5f;
                float bsq = rintf(bs[m] * QF) * INVQ / (css * 0.5f);
                float tr = ((float)aS[v] * INVQ2 + bsq) * css * 0.5f;
                dstb[(size_t)m * NPB + n] = fminf(fmaxf(t3 + tr, 0.0f), 6.0f);
            }
        }
    }
}

extern "C" void kernel_launch(void* const* d_in, const int* in_sizes, int n_in,
                              void* d_out, int out_size, void* d_ws, size_t ws_size,
                              hipStream_t stream) {
    (void)in_sizes; (void)n_in; (void)out_size; (void)ws_size;
    const float* x  = (const float*)d_in[0];
    const float* w1 = (const float*)d_in[1];
    const float* b1 = (const float*)d_in[2];
    const float* w2 = (const float*)d_in[3];
    const float* b2 = (const float*)d_in[4];
    const float* w3 = (const float*)d_in[5];
    const float* b3 = (const float*)d_in[6];
    const float* ws = (const float*)d_in[7];
    const float* bs = (const float*)d_in[8];
    float* out = (float*)d_out;

    char* base = (char*)d_ws;
    float* scales     = (float*)base;
    signed char* w1q  = (signed char*)(base + 256);
    signed char* w2q  = w1q + (size_t)256 * 512;
    signed char* w3q  = w2q + (size_t)256 * 2304;
    signed char* wsq  = w3q + (size_t)1024 * 256;
    signed char* a3q  = wsq + (size_t)1024 * 512;
    signed char* xq   = a3q + (size_t)64 * 256 * 196;
    signed char* a2q  = xq  + (size_t)64 * 512 * 784;

    const int nW1 = 256 * 512, nW2 = 256 * 2304, nW3 = 1024 * 256, nWS = 1024 * 512;
    const int nX  = 64 * 512 * 784;

    init_scales_kernel<<<1, 32, 0, stream>>>(scales);
    absmax_kernel<<<256, 256, 0, stream>>>(w1, nW1, (unsigned int*)(scales + 0));
    absmax_kernel<<<256, 256, 0, stream>>>(w2, nW2, (unsigned int*)(scales + 1));
    absmax_kernel<<<256, 256, 0, stream>>>(w3, nW3, (unsigned int*)(scales + 2));
    absmax_kernel<<<256, 256, 0, stream>>>(ws, nWS, (unsigned int*)(scales + 3));

    quantw_kernel<<<(nW1/4 + 255) / 256, 256, 0, stream>>>(w1, scales + 0, (unsigned*)w1q, nW1/4);
    quantw_kernel<<<(nW2/4 + 255) / 256, 256, 0, stream>>>(w2, scales + 1, (unsigned*)w2q, nW2/4);
    quantw_kernel<<<(nW3/4 + 255) / 256, 256, 0, stream>>>(w3, scales + 2, (unsigned*)w3q, nW3/4);
    quantw_kernel<<<(nWS/4 + 255) / 256, 256, 0, stream>>>(ws, scales + 3, (unsigned*)wsq, nWS/4);
    quantx_kernel<<<(nX/4  + 255) / 256, 256, 0, stream>>>(x, (unsigned*)xq, nX/4);

    conv1_kernel<<<dim3(13, 4, 64), 256, 0, stream>>>(xq, w1q, b1, scales, a2q);
    conv2_kernel<<<dim3(4, 4, 64), 256, 0, stream>>>(a2q, w2q, b2, scales, a3q);
    conv3s_kernel<<<dim3(4, 16, 64), 256, 0, stream>>>(a3q, xq, w3q, wsq, b3, bs, scales, out);
}
